// ContrastiveLearning_13005160973089
// MI455X (gfx1250) — matrix operand order repaired and checked by reading, not compiled, not run on hardware
//
#include <hip/hip_runtime.h>
#include <hip/hip_bf16.h>
#include <math.h>

typedef __attribute__((ext_vector_type(16))) _Float16 v16h;
typedef __attribute__((ext_vector_type(8)))  _Float16 v8h;
typedef __attribute__((ext_vector_type(4)))  _Float16 v4h;
typedef __attribute__((ext_vector_type(8)))  float    v8f;

#define DIM 128
#define TILE 16
#define MTILES 2         // row tiles per wave (M = 32 rows per wave)
#define WAVES 4          // waves per block in the main kernel
#define CHUNK_TILES 64   // 16-wide column tiles per chunk (1024 columns)

union AFrag { v16h v; v8h h[2]; };

// ---------------------------------------------------------------------------
// Kernel 1: L2-normalize rows, convert to fp16. One wave (32 lanes) per row,
// each lane handles 4 contiguous floats (128/32 = 4).
// ---------------------------------------------------------------------------
__global__ __launch_bounds__(256)
void k_normalize(const float* __restrict__ emb, _Float16* __restrict__ out, int B) {
    const int wave = threadIdx.x >> 5;
    const int lane = threadIdx.x & 31;
    const int row  = blockIdx.x * 8 + wave;
    if (row >= B) return;

    const float4 x = *(const float4*)(emb + (size_t)row * DIM + lane * 4);
    float ss = x.x * x.x + x.y * x.y + x.z * x.z + x.w * x.w;
    #pragma unroll
    for (int m = 16; m >= 1; m >>= 1) ss += __shfl_xor(ss, m, 32);

    const float s = 1.0f / fmaxf(sqrtf(ss), 1e-12f);
    v4h o;
    o.x = (_Float16)(x.x * s);
    o.y = (_Float16)(x.y * s);
    o.z = (_Float16)(x.z * s);
    o.w = (_Float16)(x.w * s);
    *(v4h*)(out + (size_t)row * DIM + lane * 4) = o;
}

// ---------------------------------------------------------------------------
// Kernel 2: fused Gram-matrix + masked exp row-sums.
// Each wave owns TWO 16-row tiles (M=32); all A fragments (32 rows x K=128)
// cached in registers. Each B-fragment load feeds two independent WMMA
// accumulator chains (8 WMMAs per 8x b128 loads). Row max is analytically
// 1/T (diagonal of a Gram matrix of unit rows), so
// exp(sim - max) = exp2((dot - 1) * 10 * log2(e)). Postprocess is fully
// branchless so the exp/cndmask/add VALU work co-executes with the XDL pipe.
// ---------------------------------------------------------------------------
__global__ __launch_bounds__(WAVES * 32)
void k_simloss(const _Float16* __restrict__ E, const int* __restrict__ labels,
               float* __restrict__ pPos, float* __restrict__ pAll, int B) {
    const int wave = threadIdx.x >> 5;
    const int lane = threadIdx.x & 31;
    const int hi   = lane >> 4;     // which 16-lane half
    const int l15  = lane & 15;
    const int rowBase0 = (blockIdx.x * WAVES + wave) * (TILE * MTILES);
    const int rowBase1 = rowBase0 + TILE;
    const int ctStart  = blockIdx.y * CHUNK_TILES;

    const float LOG2E_T = 14.426950408889634f;   // 10 * log2(e)

    // A fragment layout (16-bit A 16x32): lane L holds row M=L&15; halves e:
    // k = kb + hi*8 + (e&7) for e<8, and +16 more for e>=8. Two 16B loads.
    AFrag a0[4], a1[4];
    const _Float16* arow0 = E + (size_t)(rowBase0 + l15) * DIM;
    const _Float16* arow1 = E + (size_t)(rowBase1 + l15) * DIM;
    #pragma unroll
    for (int k4 = 0; k4 < 4; ++k4) {
        const int kb = k4 * 32;
        a0[k4].h[0] = *(const v8h*)(arow0 + kb + hi * 8);
        a0[k4].h[1] = *(const v8h*)(arow0 + kb + 16 + hi * 8);
        a1[k4].h[0] = *(const v8h*)(arow1 + kb + hi * 8);
        a1[k4].h[1] = *(const v8h*)(arow1 + kb + 16 + hi * 8);
    }

    // C/D layout: lane L holds N = L&15; VGPR v holds M = v + 8*hi.
    int labRow0[8], labRow1[8];
    #pragma unroll
    for (int v = 0; v < 8; ++v) {
        labRow0[v] = labels[rowBase0 + v + 8 * hi];
        labRow1[v] = labels[rowBase1 + v + 8 * hi];
    }

    float posAcc0[8], allAcc0[8], posAcc1[8], allAcc1[8];
    #pragma unroll
    for (int v = 0; v < 8; ++v) {
        posAcc0[v] = 0.0f; allAcc0[v] = 0.0f;
        posAcc1[v] = 0.0f; allAcc1[v] = 0.0f;
    }

    for (int ct = 0; ct < CHUNK_TILES; ++ct) {
        const int colBase = (ctStart + ct) * TILE;
        const int n       = colBase + l15;
        const int labN    = labels[n];
        // B fragment (32x16): lane L holds column N=L&15; halves e -> k = kb + hi*8 + (e&7), +16 for e>=8.
        const _Float16* brow = E + (size_t)n * DIM + hi * 8;

        // Distinct b[k4] registers: loads stay in flight across WMMAs
        // (partial loadcnt waits) instead of a full drain per k-step.
        v16h b[4];
        #pragma unroll
        for (int k4 = 0; k4 < 4; ++k4) { AFrag t; t.h[0] = *(const v8h*)(brow + k4 * 32); t.h[1] = *(const v8h*)(brow + k4 * 32 + 16); b[k4] = t.v; }

        v8f c0 = {}, c1 = {};
        #pragma unroll
        for (int k4 = 0; k4 < 4; ++k4) {
            c0 = __builtin_amdgcn_wmma_f32_16x16x32_f16(
                    false, a0[k4].v, false, b[k4], (short)0, c0, false, false);
            c1 = __builtin_amdgcn_wmma_f32_16x16x32_f16(
                    false, a1[k4].v, false, b[k4], (short)0, c1, false, false);
        }

        #pragma unroll
        for (int v = 0; v < 8; ++v) {
            const int m0 = rowBase0 + v + 8 * hi;
            const int m1 = rowBase1 + v + 8 * hi;
            // exp(sim - 1/T) with sim = dot/T: exp2(dot*10*log2e - 10*log2e)
            const float e0 = __builtin_amdgcn_exp2f(fmaf(c0[v], LOG2E_T, -LOG2E_T));
            const float e1 = __builtin_amdgcn_exp2f(fmaf(c1[v], LOG2E_T, -LOG2E_T));
            const float d0 = (m0 != n) ? e0 : 0.0f;
            const float d1 = (m1 != n) ? e1 : 0.0f;
            allAcc0[v] += d0;
            allAcc1[v] += d1;
            posAcc0[v] += (labN == labRow0[v]) ? d0 : 0.0f;
            posAcc1[v] += (labN == labRow1[v]) ? d1 : 0.0f;
        }
    }

    // Reduce the 16 column-values of each row across the 16-lane group.
    #pragma unroll
    for (int v = 0; v < 8; ++v) {
        float p0 = posAcc0[v], q0 = allAcc0[v];
        float p1 = posAcc1[v], q1 = allAcc1[v];
        #pragma unroll
        for (int off = 8; off >= 1; off >>= 1) {
            p0 += __shfl_xor(p0, off, 32);
            q0 += __shfl_xor(q0, off, 32);
            p1 += __shfl_xor(p1, off, 32);
            q1 += __shfl_xor(q1, off, 32);
        }
        if (l15 == 0) {  // lane 0 -> rows base+v, lane 16 -> rows base+8+v
            const int m0 = rowBase0 + v + 8 * hi;
            const int m1 = rowBase1 + v + 8 * hi;
            pPos[(size_t)blockIdx.y * B + m0] = p0;
            pAll[(size_t)blockIdx.y * B + m0] = q0;
            pPos[(size_t)blockIdx.y * B + m1] = p1;
            pAll[(size_t)blockIdx.y * B + m1] = q1;
        }
    }
}

// ---------------------------------------------------------------------------
// Kernel 3: single-block finalize. Label histogram (validity), deterministic
// chunk-partial sums, per-row loss, block reduction to the scalar mean.
// ---------------------------------------------------------------------------
__global__ __launch_bounds__(1024)
void k_finalize(const int* __restrict__ labels, const float* __restrict__ pPos,
                const float* __restrict__ pAll, float* __restrict__ out,
                int B, int nChunk) {
    __shared__ int   hist[256];
    __shared__ float redL[32], redV[32];
    const int tid = threadIdx.x;

    for (int i = tid; i < 256; i += 1024) hist[i] = 0;
    __syncthreads();
    for (int r = tid; r < B; r += 1024) atomicAdd(&hist[labels[r] & 255], 1);
    __syncthreads();

    float lsum = 0.0f, vsum = 0.0f;
    for (int r = tid; r < B; r += 1024) {
        float pos = 0.0f, all = 0.0f;
        for (int ch = 0; ch < nChunk; ++ch) {
            pos += pPos[(size_t)ch * B + r];
            all += pAll[(size_t)ch * B + r];
        }
        const int cnt = hist[labels[r] & 255] - 1;
        const float ratio = pos / fmaxf(all, 1e-8f);
        const float l = -logf(fmaxf(ratio, 1e-8f));
        if (cnt > 0) { lsum += l; vsum += 1.0f; }
    }

    #pragma unroll
    for (int off = 16; off >= 1; off >>= 1) {
        lsum += __shfl_xor(lsum, off, 32);
        vsum += __shfl_xor(vsum, off, 32);
    }
    const int wid = tid >> 5;
    if ((tid & 31) == 0) { redL[wid] = lsum; redV[wid] = vsum; }
    __syncthreads();
    if (wid == 0) {
        lsum = redL[tid & 31];
        vsum = redV[tid & 31];
        #pragma unroll
        for (int off = 16; off >= 1; off >>= 1) {
            lsum += __shfl_xor(lsum, off, 32);
            vsum += __shfl_xor(vsum, off, 32);
        }
        if (tid == 0)
            out[0] = (vsum > 0.0f) ? (lsum / fmaxf(vsum, 1.0f)) : 0.0f;
    }
}

// ---------------------------------------------------------------------------
extern "C" void kernel_launch(void* const* d_in, const int* in_sizes, int n_in,
                              void* d_out, int out_size, void* d_ws, size_t ws_size,
                              hipStream_t stream) {
    const float* emb    = (const float*)d_in[0];
    const int*   labels = (const int*)d_in[1];
    const int    B      = in_sizes[1];                  // 8192
    const int    nChunk = B / (CHUNK_TILES * TILE);     // 8

    // Workspace layout: fp16 normalized matrix | pos partials | all partials
    _Float16* E    = (_Float16*)d_ws;
    float*    pPos = (float*)((char*)d_ws + (size_t)B * DIM * sizeof(_Float16));
    float*    pAll = pPos + (size_t)nChunk * B;

    k_normalize<<<B / 8, 256, 0, stream>>>(emb, E, B);

    dim3 grid(B / (TILE * MTILES * WAVES), nChunk);
    k_simloss<<<grid, WAVES * 32, 0, stream>>>(E, labels, pPos, pAll, B);

    k_finalize<<<1, 1024, 0, stream>>>(labels, pPos, pAll, (float*)d_out, B, nChunk);
}
